// Gaussian_Distance_KL_Jensen_19172734009625
// MI455X (gfx1250) — compile-verified, matrix operand order checked by reading, not run on hardware
//
#include <hip/hip_runtime.h>

typedef __attribute__((ext_vector_type(2)))  float    v2f;
typedef __attribute__((ext_vector_type(8)))  float    v8f;
typedef __attribute__((ext_vector_type(16))) _Float16 v16h;

#define BATCH   32
#define CH      32
#define HH      64
#define WW      64
#define BSTRIDE (CH * HH * WW) /* 131072 elements per batch index */

// Full-wave (32-lane) sum using WMMA. Returns the total in every lane.
// A-matrix (16x4 or 16x32, documented layouts): lane L<16 supplies row M=L at the
// low-K slots, lane L>=16 supplies row M=L-16 at the high-K slots. With B = all-ones
// (layout-independent), D[m][n] = val_m + val_{m+16}; summing the 8 D VGPRs gives the
// half sums in lanes 0-15 / 16-31, and one shfl_xor(16) completes the reduction.
__device__ __forceinline__ float wave_reduce_wmma(float val) {
#if __has_builtin(__builtin_amdgcn_wmma_f32_16x16x4_f32)
    v2f a; a[0] = val;  a[1] = 0.0f;
    v2f b; b[0] = 1.0f; b[1] = 1.0f;
    v8f c = {};
    v8f d = __builtin_amdgcn_wmma_f32_16x16x4_f32(
        false, a, false, b, (short)0, c, false, false);
    float s = ((d[0] + d[1]) + (d[2] + d[3])) + ((d[4] + d[5]) + (d[6] + d[7]));
    s += __shfl_xor(s, 16, 32);
    return s;
#else
    // f16 WMMA fallback with exact hi/lo split (f32-accurate; f32 accumulator).
    const float SC = 0x1p-8f, ISC = 0x1p8f;   // exact power-of-2 range guard
    float sval = val * SC;
    _Float16 hi = (_Float16)sval;
    _Float16 lo = (_Float16)(sval - (float)hi);
    v16h a;
    #pragma unroll
    for (int k = 0; k < 16; ++k) a[k] = (_Float16)0.0f;
    a[0] = hi; a[1] = lo;
    v16h b;
    #pragma unroll
    for (int k = 0; k < 16; ++k) b[k] = (_Float16)1.0f;
    v8f c = {};
    v8f d = __builtin_amdgcn_wmma_f32_16x16x32_f16(
        false, a, false, b, (short)0, c, false, false);
    float s = ((d[0] + d[1]) + (d[2] + d[3])) + ((d[4] + d[5]) + (d[6] + d[7]));
    s += __shfl_xor(s, 16, 32);
    return s * ISC;
#endif
}

// Grid: 512 blocks. bid -> c = bid>>4 (channel), hh2 = bid&15 (pair of pooled rows).
// Block: 256 threads. tx -> ww = tx&31 (pooled col, coalesced lanes),
//                          y  = tx>>5: hsub = y&1 (which pooled row), chunk = y>>2? no: y>>1 (batch chunk of 8).
__global__ __launch_bounds__(256) void gkl_main(
    const float* __restrict__ mu_a, const float* __restrict__ lv_a,
    const float* __restrict__ mu_b, const float* __restrict__ lv_b,
    float* __restrict__ partials) {
    const int tx    = threadIdx.x;
    const int ww    = tx & 31;
    const int y     = tx >> 5;        // 0..7
    const int hsub  = y & 1;          // pooled-row within pair
    const int chunk = y >> 1;         // 0..3 : batch indices [8*chunk, 8*chunk+8)
    const int bid   = blockIdx.x;
    const int c     = bid >> 4;       // 0..31
    const int hh    = (bid & 15) * 2 + hsub;  // pooled row 0..31
    const int h0    = hh * 2;
    const int w0    = ww * 2;

    int base = ((chunk * 8 * CH + c) * HH + h0) * WW + w0;

    const float INV_S2PI = 0.3989422804014327f; // 1/sqrt(2*pi)
    float A1 = 0.f, A2 = 0.f, A3 = 0.f;
    float Ra0 = 0.f, Ra1 = 0.f, Ra2 = 0.f, La = 0.f;
    float Rb0 = 0.f, Rb1 = 0.f, Rb2 = 0.f, Lb = 0.f;

    #pragma unroll
    for (int ii = 0; ii < 8; ++ii) {
        const float2 ma01 = *(const float2*)(mu_a + base);
        const float2 ma23 = *(const float2*)(mu_a + base + WW);
        const float2 la01 = *(const float2*)(lv_a + base);
        const float2 la23 = *(const float2*)(lv_a + base + WW);
        const float2 mb01 = *(const float2*)(mu_b + base);
        const float2 mb23 = *(const float2*)(mu_b + base + WW);
        const float2 lb01 = *(const float2*)(lv_b + base);
        const float2 lb23 = *(const float2*)(lv_b + base + WW);
        base += BSTRIDE;

        float ma = 0.25f   * ((ma01.x + ma01.y) + (ma23.x + ma23.y));
        float mb = 0.25f   * ((mb01.x + mb01.y) + (mb23.x + mb23.y));
        float va = 0.0625f * ((__expf(la01.x) + __expf(la01.y)) +
                              (__expf(la23.x) + __expf(la23.y)));
        float vb = 0.0625f * ((__expf(lb01.x) + __expf(lb01.y)) +
                              (__expf(lb23.x) + __expf(lb23.y)));

        float rva = __builtin_amdgcn_rcpf(va);
        float rvb = __builtin_amdgcn_rcpf(vb);
        float ia  = 0.5f * rva * rva;   // 1/(2 va^2)
        float ib  = 0.5f * rvb * rvb;   // 1/(2 vb^2)

        A1 += va * va;  A2 += ma;  A3 += ma * ma;
        Ra0 += ia;  Ra1 += ma * ia;  Ra2 += (ma * ma) * ia;
        La  += __logf(__builtin_fmaf(rva, INV_S2PI, 1e-6f));
        Rb0 += ib;  Rb1 += mb * ib;  Rb2 += (mb * mb) * ib;
        Lb  += __logf(__builtin_fmaf(rvb, INV_S2PI, 1e-6f));
    }

    // Combine the 4 batch-chunks per pixel (all 11 stats are additive over batch).
    __shared__ float red[8][32][12];
    __shared__ float sS[64];
    red[y][ww][0] = A1;  red[y][ww][1] = A2;  red[y][ww][2] = A3;
    red[y][ww][3] = Ra0; red[y][ww][4] = Ra1; red[y][ww][5] = Ra2; red[y][ww][6] = La;
    red[y][ww][7] = Rb0; red[y][ww][8] = Rb1; red[y][ww][9] = Rb2; red[y][ww][10] = Lb;
    __syncthreads();

    if (y < 2) {
        float t[11];
        #pragma unroll
        for (int q = 0; q < 11; ++q)
            t[q] = (red[y][ww][q] + red[y + 2][ww][q]) +
                   (red[y + 4][ww][q] + red[y + 6][ww][q]);
        float AA = t[0] + t[2];   // A1 + A3
        float S  = 32.0f * (t[6] - t[10])
                 - (AA * t[3] - 2.0f * t[1] * t[4] + 32.0f * t[5])
                 + (AA * t[7] - 2.0f * t[1] * t[8] + 32.0f * t[9]);
        sS[y * 32 + ww] = S;
    }
    __syncthreads();

    if (tx < 32) {                       // wave 0 only: EXEC all-ones for WMMA
        float v   = sS[tx] + sS[tx + 32];
        float tot = wave_reduce_wmma(v);
        if (tx == 0) partials[bid] = tot;
    }
}

__global__ __launch_bounds__(32) void gkl_finish(
    const float* __restrict__ partials, float* __restrict__ out) {
    const int lane = threadIdx.x;        // exactly one full wave
    float v = 0.f;
    #pragma unroll
    for (int k = 0; k < 16; ++k) v += partials[lane + 32 * k];
    float tot = wave_reduce_wmma(v);
    if (lane == 0) out[0] = tot * (1.0f / 1024.0f);   // / (B*B)
}

extern "C" void kernel_launch(void* const* d_in, const int* in_sizes, int n_in,
                              void* d_out, int out_size, void* d_ws, size_t ws_size,
                              hipStream_t stream) {
    const float* mu_a = (const float*)d_in[0];
    const float* lv_a = (const float*)d_in[1];
    const float* mu_b = (const float*)d_in[2];
    const float* lv_b = (const float*)d_in[3];
    float* partials = (float*)d_ws;      // 512 floats, fully overwritten each call

    gkl_main<<<512, 256, 0, stream>>>(mu_a, lv_a, mu_b, lv_b, partials);
    gkl_finish<<<1, 32, 0, stream>>>(partials, (float*)d_out);
}
